// PurePythonLSTM_6330781794634
// MI455X (gfx1250) — matrix operand order repaired and checked by reading, not compiled, not run on hardware
//
#include <hip/hip_runtime.h>

typedef __attribute__((ext_vector_type(16))) _Float16 v16h;
typedef __attribute__((ext_vector_type(8)))  _Float16 v8h;
typedef __attribute__((ext_vector_type(8)))  float    v8f;

union U16 { v16h v; v8h h[2]; };

#define VOCAB 287
#define VPAD  288     // vocab rows padded (row 287 = zeros)
#define HID   18
#define KP    32      // padded K stride (halfs) for all f16 rows
#define BB    256
#define TT    512
#define NGC   72      // 4 gates * 18 concatenated output cols
#define NT_G  5       // ceil(72/16)
#define NT_V  18      // ceil(287/16)
#define VWAVES 8
#define MTPW   4      // M-tiles per wave in vocab kernel

__device__ __forceinline__ float rcpf(float x) { return __builtin_amdgcn_rcpf(x); }
__device__ __forceinline__ float sigf(float x) { return rcpf(1.0f + __expf(-x)); }
// branch-free tanh: no range-test branches, saturates cleanly for large |x|
__device__ __forceinline__ float tanh_fast(float x) {
  const float e = __expf(-2.0f * fabsf(x));
  const float r = (1.0f - e) * rcpf(1.0f + e);
  return copysignf(r, x);
}

// f16 16x16x32 A fragment: element j holds K = j + 16*(lane>=16)
// => per lane: one contiguous 16-half run at 2*kb, kb = 8*(lane>=16).
// With rows padded to KP=32 halfs this is two aligned 16B loads.
__device__ __forceinline__ v16h load_a_frag(const _Float16* rowbase, int kb) {
  const v8h* p = (const v8h*)(rowbase + 2 * kb);
  U16 u; u.h[0] = p[0]; u.h[1] = p[1];
  return u.v;
}
// f16 32x16 B fragment: element j holds K = j + 16*(lane>=16): one 16-half run.
__device__ __forceinline__ v16h load_b_frag(const _Float16* rowbase, int kb16) {
  const v8h* p = (const v8h*)(rowbase + kb16);
  U16 u; u.h[0] = p[0]; u.h[1] = p[1];
  return u.v;
}

// Bias trick: every A row carries 1.0 at K=18 (first pad slot); bias vectors
// live in row K=18 of the corresponding B-tiles. Accumulators then start from
// inline 0 (no per-iteration bias-splat movs). The recurrent B-tiles keep
// zeros at K=18, so the 1.0 in the h fragment is inert there.

// ---------------------------------------------------------------------------
// Kernel 1: fused embedding gather + input projection + LSTM scan.
// 16 blocks (B/16) x 1 wave32. Weight B-tiles stationary in VGPRs; embedding
// table staged in LDS as padded f16 rows so every per-step tile build is
// two ds_load_b128. h history (padded rows) copied out with b128 stores.
// ---------------------------------------------------------------------------
__global__ __launch_bounds__(32)
void lstm_scan_kernel(const int* __restrict__ x, const float* __restrict__ emb,
                      const float* __restrict__ W_ii, const float* __restrict__ b_ii,
                      const float* __restrict__ W_hi,
                      const float* __restrict__ W_if, const float* __restrict__ b_if,
                      const float* __restrict__ W_hf,
                      const float* __restrict__ W_ig, const float* __restrict__ b_ig,
                      const float* __restrict__ W_hg,
                      const float* __restrict__ W_io, const float* __restrict__ b_io,
                      const float* __restrict__ W_ho,
                      _Float16* __restrict__ hs) {
  __shared__ _Float16 embLDS[VOCAB * KP];   // 18 KB; K=18 -> 1.0, K>18 -> 0
  __shared__ _Float16 hLDS[16 * KP];        // h tile; K=18 -> 1.0, K>18 -> 0
  __shared__ float    preLDS[16 * 80];      // gate preacts [M][Ncombined pad 80]

  const int lane = threadIdx.x;
  const int mrow = lane & 15;
  const int b0   = blockIdx.x * 16;
  const int mhi  = (lane & 16) ? 8 : 0;
  const int kbA  = (lane & 16) ? 8 : 0;     // A fragment first-run base

  // ---- one-time: stage padded f16 embedding table + init h tile ----
  for (int i = lane; i < VOCAB * KP; i += 32) {
    const int row = i >> 5, k = i & 31;
    embLDS[i] = (k < HID)  ? (_Float16)emb[row * HID + k]
              : (k == HID) ? (_Float16)1.0f : (_Float16)0.0f;
  }
  for (int i = lane; i < 16 * KP; i += 32)
    hLDS[i] = ((i & 31) == HID) ? (_Float16)1.0f : (_Float16)0.0f;

  // ---- stationary weight B-tiles; bias folded into Wx row K=18 ----
  const float* Wx[4] = { W_ii, W_if, W_ig, W_io };
  const float* Wh[4] = { W_hi, W_hf, W_hg, W_ho };
  const float* bv[4] = { b_ii, b_if, b_ig, b_io };
  v16h wxT[NT_G], whT[NT_G];
  #pragma unroll
  for (int nt = 0; nt < NT_G; ++nt) {
    const int n = nt * 16 + mrow;
    const int gate = (n < NGC) ? (n / HID) : 0;
    const int gcol = (n < NGC) ? (n % HID) : 0;
    #pragma unroll
    for (int j = 0; j < 16; ++j) {
      const int K = j + ((lane & 16) ? 16 : 0);
      float wx = 0.0f, wh = 0.0f;
      if (n < NGC) {
        if (K < HID) {                     // out = in @ W.T => B(K,N) = W[gcol,K]
          wx = Wx[gate][gcol * HID + K];
          wh = Wh[gate][gcol * HID + K];
        } else if (K == HID) {
          wx = bv[gate][gcol];             // bias row, pairs with A's 1.0
        }
      }
      wxT[nt][j] = (_Float16)wx;
      whT[nt][j] = (_Float16)wh;
    }
  }
  __syncthreads();

  float c[9];
  #pragma unroll
  for (int q = 0; q < 9; ++q) c[q] = 0.0f;

  // h-tile writeback assignment: each lane copies two 16B chunks per step
  const int cpm = lane >> 2;                // row 0..7 (and +8)
  const int cpk = (lane & 3) * 8;           // half offset within padded row

  // embedding fragment for t=0 (software-pipelined across the loop)
  v16h ae = load_a_frag(embLDS + x[(b0 + mrow) * TT + 0] * KP, kbA);

  #pragma unroll 1
  for (int t = 0; t < TT; ++t) {
    const v16h ah = load_a_frag(hLDS + mrow * KP, kbA);

    #pragma unroll
    for (int nt = 0; nt < NT_G; ++nt) {
      v8f acc = {};                         // inline 0 C operand
      acc = __builtin_amdgcn_wmma_f32_16x16x32_f16(false, ae, false, wxT[nt],
                                                   (short)0, acc, false, false);
      acc = __builtin_amdgcn_wmma_f32_16x16x32_f16(false, ah, false, whT[nt],
                                                   (short)0, acc, false, false);
      #pragma unroll
      for (int r = 0; r < 8; ++r)
        preLDS[(r + mhi) * 80 + nt * 16 + mrow] = acc[r];
    }

    // prefetch next step's embedding fragment (independent of this step)
    const int tn = (t + 1 < TT) ? (t + 1) : t;   // uniform clamp
    const v16h aen = load_a_frag(embLDS + x[(b0 + mrow) * TT + tn] * KP, kbA);

    __syncthreads();

    // Gate fusion: 288 = 16 rows * 18 cols, 9 per lane, fixed (m,jc) map.
    #pragma unroll
    for (int q = 0; q < 9; ++q) {
      const int idx = lane + q * 32;
      const int m  = idx / HID;
      const int jc = idx % HID;
      const float it = sigf     (preLDS[m * 80 +           jc]);
      const float ft = sigf     (preLDS[m * 80 +     HID + jc]);
      const float gt = tanh_fast(preLDS[m * 80 + 2 * HID + jc]);
      const float ot = sigf     (preLDS[m * 80 + 3 * HID + jc]);
      c[q] = ft * c[q] + it * gt;
      const float hv = ot * tanh_fast(c[q]);
      hLDS[m * KP + jc] = (_Float16)hv;    // pads (jc>=18) never touched
    }
    __syncthreads();

    // Vectorized h-history writeback: whole padded tile (incl. 1.0 at K=18),
    // 2 x (ds_load_b128 + global_store_b128) per lane.
    {
      const v8h lo = *(const v8h*)(hLDS +  cpm      * KP + cpk);
      const v8h hi = *(const v8h*)(hLDS + (cpm + 8) * KP + cpk);
      *(v8h*)(hs + ((size_t)(b0 + cpm)     * TT + t) * KP + cpk) = lo;
      *(v8h*)(hs + ((size_t)(b0 + cpm + 8) * TT + t) * KP + cpk) = hi;
    }
    ae = aen;
  }
}

// ---------------------------------------------------------------------------
// Kernel 2: vocab projection  out[row, v] = hs[row,:] @ W_vocab[v,:] + b[v].
// 256 threads (8 waves); W_vocab (with bias folded into K=18 row) staged in
// LDS once per block; each wave owns 4 M-tiles, loops 18 N-tiles.
// ---------------------------------------------------------------------------
__global__ __launch_bounds__(VWAVES * 32)
void vocab_kernel(const _Float16* __restrict__ hs,
                  const float* __restrict__ W_vocab,
                  const float* __restrict__ b_vocab,
                  float* __restrict__ out) {
  __shared__ _Float16 wLDS[VPAD * KP];      // 18 KB, padded rows/K

  const int tid  = threadIdx.x;
  const int lane = tid & 31;
  const int wave = tid >> 5;
  const int mrow = lane & 15;
  const int mhi  = (lane & 16) ? 8 : 0;
  const int kbA  = (lane & 16) ? 8 : 0;
  const int kbB  = (lane & 16) ? 16 : 0;

  for (int i = tid; i < VPAD * KP; i += VWAVES * 32) {
    const int row = i >> 5, k = i & 31;
    float w = 0.0f;
    if (row < VOCAB) {
      if (k < HID)       w = W_vocab[row * HID + k];
      else if (k == HID) w = b_vocab[row];   // pairs with A's 1.0 at K=18
    }
    wLDS[i] = (_Float16)w;
  }
  __syncthreads();

  #pragma unroll 1
  for (int mi = 0; mi < MTPW; ++mi) {
    const size_t mtile = (size_t)blockIdx.x * (VWAVES * MTPW) + wave * MTPW + mi;
    const size_t row0  = mtile * 16;
    // A fragment: two aligned global_load_b128 from padded hs rows
    const v16h a = load_a_frag(hs + (row0 + mrow) * KP, kbA);

    #pragma unroll 1
    for (int nt = 0; nt < NT_V; ++nt) {
      const int n = nt * 16 + mrow;                    // < VPAD, unconditional
      const v16h bt = load_b_frag(wLDS + n * KP, kbB); // two ds_load_b128
      v8f acc = {};                                    // inline 0 C operand
      acc = __builtin_amdgcn_wmma_f32_16x16x32_f16(false, a, false, bt,
                                                   (short)0, acc, false, false);
      if (n < VOCAB) {
        #pragma unroll
        for (int r = 0; r < 8; ++r)
          out[(row0 + r + mhi) * VOCAB + n] = acc[r];
      }
    }
  }
}

// ---------------------------------------------------------------------------
extern "C" void kernel_launch(void* const* d_in, const int* in_sizes, int n_in,
                              void* d_out, int out_size, void* d_ws, size_t ws_size,
                              hipStream_t stream) {
  (void)in_sizes; (void)n_in; (void)out_size; (void)ws_size;
  const int*   x       = (const int*)  d_in[0];
  const float* emb     = (const float*)d_in[1];
  const float* W_ii    = (const float*)d_in[2];
  const float* b_ii    = (const float*)d_in[3];
  const float* W_hi    = (const float*)d_in[4];
  const float* W_if    = (const float*)d_in[5];
  const float* b_if    = (const float*)d_in[6];
  const float* W_hf    = (const float*)d_in[7];
  const float* W_ig    = (const float*)d_in[8];
  const float* b_ig    = (const float*)d_in[9];
  const float* W_hg    = (const float*)d_in[10];
  const float* W_io    = (const float*)d_in[11];
  const float* b_io    = (const float*)d_in[12];
  const float* W_ho    = (const float*)d_in[13];
  const float* W_vocab = (const float*)d_in[14];
  const float* b_vocab = (const float*)d_in[15];
  float*    out = (float*)d_out;
  _Float16* hs  = (_Float16*)d_ws;   // B*T rows x KP halfs = 8.4 MB, fully
                                     // rewritten (incl. pads) every call

  lstm_scan_kernel<<<BB / 16, 32, 0, stream>>>(x, emb,
      W_ii, b_ii, W_hi, W_if, b_if, W_hf, W_ig, b_ig, W_hg, W_io, b_io, W_ho, hs);

  vocab_kernel<<<(BB * TT / 16) / (VWAVES * MTPW), VWAVES * 32, 0, stream>>>(
      hs, W_vocab, b_vocab, out);
}
